// DynamicOversizeConv2d_1906965480147
// MI455X (gfx1250) — matrix operand order repaired and checked by reading, not compiled, not run on hardware
//
#include <hip/hip_runtime.h>

typedef __attribute__((ext_vector_type(16))) __bf16 v16bf;
typedef __attribute__((ext_vector_type(8)))  float  v8f;

#define N96   96
#define LDB   100   // bf16 LDS row stride (elements)
#define LDSF  100   // f32 score LDS row stride (elements)
#define NTHR  384   // 12 waves

// ---- WMMA fragment loaders (this code's contiguous k order k = 16h + s for A and B, wave32) -------------------

// A-matrix 16x32 bf16, source row-major [m][k]:
// lanes 0-15: M=lane, VGPR v holds K = 2v , +1
// lanes 16-31: same M, K offset +16
__device__ inline v16bf load_a_frag(const __bf16* src, int ld, int m0, int k0, int lane) {
  const int m = m0 + (lane & 15);
  const int kh = (lane < 16) ? 0 : 16;
  v16bf f;
#pragma unroll
  for (int v = 0; v < 8; ++v) {
    int k = k0 + 2 * v + kh;
    f[2 * v]     = src[m * ld + k];
    f[2 * v + 1] = src[m * ld + k + 1];
  }
  return f;
}

// B-matrix 32x16 bf16 where element B(k,n) = srcT[n][k] (contiguous in k):
// lanes 0-15: N=lane, K=0..15 across VGPR0..7 (2 per VGPR); lanes 16-31: K=16..31
__device__ inline v16bf load_b_fragT(const __bf16* srcT, int ld, int n0, int k0, int lane) {
  const int n  = n0 + (lane & 15);
  const int kb = k0 + ((lane < 16) ? 0 : 16);
  v16bf f;
#pragma unroll
  for (int v = 0; v < 8; ++v) {
    int k = kb + 2 * v;
    f[2 * v]     = srcT[n * ld + k];
    f[2 * v + 1] = srcT[n * ld + k + 1];
  }
  return f;
}

// B-matrix 32x16 bf16 where element B(k,n) = src[k][n] (strided in k)
__device__ inline v16bf load_b_fragN(const __bf16* src, int ld, int n0, int k0, int lane) {
  const int n  = n0 + (lane & 15);
  const int kb = k0 + ((lane < 16) ? 0 : 16);
  v16bf f;
#pragma unroll
  for (int v = 0; v < 8; ++v) {
    int k = kb + 2 * v;
    f[2 * v]     = src[k * ld + n];
    f[2 * v + 1] = src[(k + 1) * ld + n];
  }
  return f;
}

__device__ inline v8f wmma_bf16(v16bf a, v16bf b, v8f c) {
  return __builtin_amdgcn_wmma_f32_16x16x32_bf16(false, a, false, b, (short)0, c,
                                                 false, false);
}

// ---------------------------------------------------------------------------
// One workgroup handles one (b,c) pair: full 96x96 axial-attention pipeline.
//   S_h = Q K^T ; A_h = softmax_row(S_h) + band_h
//   V1  = A_h V
//   S_w = Q^T K ; A_w = softmax_row(S_w) + band_w
//   Out = V1 A_w^T            (already [h][w] order)
// ---------------------------------------------------------------------------
__global__ __launch_bounds__(NTHR, 1)
void axial_attn_band_kernel(const float* __restrict__ qg,
                            const float* __restrict__ kg,
                            const float* __restrict__ vg,
                            const float* __restrict__ kerh,
                            const float* __restrict__ kerw,
                            float* __restrict__ outg, int C) {
  __shared__ __bf16 sQ [N96 * LDB];  // Q row-major; reused as V1 after GEMM2
  __shared__ __bf16 sQt[N96 * LDB];  // Q transposed
  __shared__ __bf16 sK [N96 * LDB];  // K row-major
  __shared__ __bf16 sVt[N96 * LDB];  // V transposed
  __shared__ __bf16 sA [N96 * LDB];  // attention matrix (bf16), reused
  __shared__ float  sS [N96 * LDSF]; // f32 scores
  __shared__ float  sKer[2][16];     // band kernels (13 taps each)

  const int tid  = threadIdx.x;
  const int bc   = blockIdx.x;
  const int c    = bc % C;
  const size_t base = (size_t)bc * (N96 * N96);

  if (tid < 13) {
    sKer[0][tid] = kerh[c * 13 + tid];
    sKer[1][tid] = kerw[c * 13 + tid];
  }

  // ---- stage Q, K, V into LDS as bf16 (vectorized b128 global loads) ----
  for (int i = tid; i < (N96 * N96) / 4; i += NTHR) {
    const int h = (i * 4) / N96;
    const int w = (i * 4) % N96;
    float4 fq = ((const float4*)(qg + base))[i];
    float4 fk = ((const float4*)(kg + base))[i];
    float4 fv = ((const float4*)(vg + base))[i];
    sQ[h * LDB + w + 0] = (__bf16)fq.x;  sQ[h * LDB + w + 1] = (__bf16)fq.y;
    sQ[h * LDB + w + 2] = (__bf16)fq.z;  sQ[h * LDB + w + 3] = (__bf16)fq.w;
    sQt[(w + 0) * LDB + h] = (__bf16)fq.x;  sQt[(w + 1) * LDB + h] = (__bf16)fq.y;
    sQt[(w + 2) * LDB + h] = (__bf16)fq.z;  sQt[(w + 3) * LDB + h] = (__bf16)fq.w;
    sK[h * LDB + w + 0] = (__bf16)fk.x;  sK[h * LDB + w + 1] = (__bf16)fk.y;
    sK[h * LDB + w + 2] = (__bf16)fk.z;  sK[h * LDB + w + 3] = (__bf16)fk.w;
    sVt[(w + 0) * LDB + h] = (__bf16)fv.x;  sVt[(w + 1) * LDB + h] = (__bf16)fv.y;
    sVt[(w + 2) * LDB + h] = (__bf16)fv.z;  sVt[(w + 3) * LDB + h] = (__bf16)fv.w;
  }
  __syncthreads();

  const int wave = tid >> 5;
  const int lane = tid & 31;
  const int mrow = (lane >> 4) * 8;   // D-frag: M = r + mrow
  const int ncol = lane & 15;         // D-frag: N = ncol

  // ---- GEMM1: S_h = Q K^T  (B(k,n)=K[n][k] -> contiguous from sK) ----
#pragma unroll
  for (int t = 0; t < 3; ++t) {
    const int tile = wave + 12 * t;
    const int tm = (tile / 6) * 16, tn = (tile % 6) * 16;
    v8f acc = {};
#pragma unroll
    for (int kk = 0; kk < N96; kk += 32) {
      v16bf a = load_a_frag (sQ, LDB, tm, kk, lane);
      v16bf b = load_b_fragT(sK, LDB, tn, kk, lane);
      acc = wmma_bf16(a, b, acc);
    }
#pragma unroll
    for (int r = 0; r < 8; ++r)
      sS[(tm + r + mrow) * LDSF + tn + ncol] = acc[r];
  }
  __syncthreads();

  // ---- softmax rows of S + band_h -> sA (bf16); 4 lanes per row ----
  {
    const int row = tid >> 2, sub = tid & 3;
    float vals[24];
    float mx = -3.0e38f;
#pragma unroll
    for (int t = 0; t < 24; ++t) {
      vals[t] = sS[row * LDSF + sub + 4 * t];
      mx = fmaxf(mx, vals[t]);
    }
    mx = fmaxf(mx, __shfl_xor(mx, 1, 4));
    mx = fmaxf(mx, __shfl_xor(mx, 2, 4));
    float sum = 0.f;
#pragma unroll
    for (int t = 0; t < 24; ++t) { vals[t] = __expf(vals[t] - mx); sum += vals[t]; }
    sum += __shfl_xor(sum, 1, 4);
    sum += __shfl_xor(sum, 2, 4);
    const float inv = 1.f / sum;
#pragma unroll
    for (int t = 0; t < 24; ++t) {
      const int j = sub + 4 * t;
      const int off = j - row + 6;
      const float band = (off >= 0 && off < 13) ? sKer[0][off] : 0.f;
      sA[row * LDB + j] = (__bf16)(vals[t] * inv + band);
    }
  }
  __syncthreads();

  // ---- GEMM2: V1 = A_h V  (B(k,n)=V[k][n]=sVt[n][k]) -> V1 into sQ ----
  // ---- GEMM3: S_w = Q^T K (A from sQt; B(k,n)=K[k][n] strided)      ----
#pragma unroll
  for (int t = 0; t < 3; ++t) {
    const int tile = wave + 12 * t;
    const int tm = (tile / 6) * 16, tn = (tile % 6) * 16;
    v8f acc = {};
#pragma unroll
    for (int kk = 0; kk < N96; kk += 32) {
      v16bf a = load_a_frag (sA,  LDB, tm, kk, lane);
      v16bf b = load_b_fragT(sVt, LDB, tn, kk, lane);
      acc = wmma_bf16(a, b, acc);
    }
#pragma unroll
    for (int r = 0; r < 8; ++r)
      sQ[(tm + r + mrow) * LDB + tn + ncol] = (__bf16)acc[r];   // V1

    v8f acc2 = {};
#pragma unroll
    for (int kk = 0; kk < N96; kk += 32) {
      v16bf a = load_a_frag (sQt, LDB, tm, kk, lane);
      v16bf b = load_b_fragN(sK,  LDB, tn, kk, lane);
      acc2 = wmma_bf16(a, b, acc2);
    }
#pragma unroll
    for (int r = 0; r < 8; ++r)
      sS[(tm + r + mrow) * LDSF + tn + ncol] = acc2[r];         // S_w
  }
  __syncthreads();

  // ---- softmax rows of S_w + band_w -> sA (bf16) ----
  {
    const int row = tid >> 2, sub = tid & 3;
    float vals[24];
    float mx = -3.0e38f;
#pragma unroll
    for (int t = 0; t < 24; ++t) {
      vals[t] = sS[row * LDSF + sub + 4 * t];
      mx = fmaxf(mx, vals[t]);
    }
    mx = fmaxf(mx, __shfl_xor(mx, 1, 4));
    mx = fmaxf(mx, __shfl_xor(mx, 2, 4));
    float sum = 0.f;
#pragma unroll
    for (int t = 0; t < 24; ++t) { vals[t] = __expf(vals[t] - mx); sum += vals[t]; }
    sum += __shfl_xor(sum, 1, 4);
    sum += __shfl_xor(sum, 2, 4);
    const float inv = 1.f / sum;
#pragma unroll
    for (int t = 0; t < 24; ++t) {
      const int j = sub + 4 * t;
      const int off = j - row + 6;
      const float band = (off >= 0 && off < 13) ? sKer[1][off] : 0.f;
      sA[row * LDB + j] = (__bf16)(vals[t] * inv + band);
    }
  }
  __syncthreads();

  // ---- GEMM4: Out = V1 A_w^T  (B(k,n)=A_w[n][k] contiguous from sA) ----
#pragma unroll
  for (int t = 0; t < 3; ++t) {
    const int tile = wave + 12 * t;
    const int tm = (tile / 6) * 16, tn = (tile % 6) * 16;
    v8f acc = {};
#pragma unroll
    for (int kk = 0; kk < N96; kk += 32) {
      v16bf a = load_a_frag (sQ, LDB, tm, kk, lane);   // V1
      v16bf b = load_b_fragT(sA, LDB, tn, kk, lane);   // A_w^T
      acc = wmma_bf16(a, b, acc);
    }
#pragma unroll
    for (int r = 0; r < 8; ++r)
      outg[base + (size_t)(tm + r + mrow) * N96 + tn + ncol] = acc[r];
  }
}

extern "C" void kernel_launch(void* const* d_in, const int* in_sizes, int n_in,
                              void* d_out, int out_size, void* d_ws, size_t ws_size,
                              hipStream_t stream) {
  const float* q  = (const float*)d_in[0];
  const float* k  = (const float*)d_in[1];
  const float* v  = (const float*)d_in[2];
  const float* kh = (const float*)d_in[3];
  const float* kw = (const float*)d_in[4];
  float* out = (float*)d_out;

  const int C  = in_sizes[3] / 13;             // kernel_h is (C, 13)
  const int BC = in_sizes[0] / (N96 * N96);    // B*C pairs

  hipLaunchKernelGGL(axial_attn_band_kernel, dim3(BC), dim3(NTHR), 0, stream,
                     q, k, v, kh, kw, out, C);
}
